// Global2PointGraph_48524540510935
// MI455X (gfx1250) — matrix operand order repaired and checked by reading, not compiled, not run on hardware
//
#include <hip/hip_runtime.h>
#include <hip/hip_bf16.h>
#include <math.h>

// ---------------- problem constants (match setup_inputs) ----------------
#define B_    4
#define S_    24          // new_size
#define N576  576         // S_*S_
#define D_    256         // dim
#define P_    128         // nPnt
#define PH    512         // 4*P_ (MLP hidden)
#define H_    120
#define W_    160
#define HW_   19200
#define NEW2  48
#define QD    32768       // P_*D_

typedef __bf16 bf16_t;
typedef __attribute__((ext_vector_type(16))) __bf16 v16bf;
typedef __attribute__((ext_vector_type(8)))  __bf16 v8bf;
typedef __attribute__((ext_vector_type(8)))  float  v8f;

// ---------------- WMMA helpers (CDNA5 wave32, 16x16x32 bf16) ----------------
__device__ __forceinline__ v8f wmma_bf16(v16bf a, v16bf b, v8f c) {
  // 8 args: (neg_a, A, neg_b, B, c_mod, C, reuse_a, reuse_b)
  return __builtin_amdgcn_wmma_f32_16x16x32_bf16(false, a, false, b, (short)0, c,
                                                 false, false);
}

__device__ __forceinline__ v16bf frag_from_chunks(const bf16_t* lo, const bf16_t* hi) {
  v8bf x = *reinterpret_cast<const v8bf*>(lo);
  v8bf y = *reinterpret_cast<const v8bf*>(hi);
  v16bf a;
#pragma unroll
  for (int j = 0; j < 8; ++j) { a[j] = x[j]; a[8 + j] = y[j]; }
  return a;
}

// A operand (16x32, row-major source): rowp already points at this lane's row
// (m = lane&15). A layout (matches B frags): lanes 0-15 hold K {k0..k0+15},
// lanes 16-31 hold K {k0+16..k0+31}.
__device__ __forceinline__ v16bf load_a_frag(const bf16_t* rowp, int k0, int lane) {
  int off = (lane < 16) ? 0 : 16;
  return frag_from_chunks(rowp + k0 + off, rowp + k0 + off + 8);
}

// B operand from pre-swizzled fragment buffer: 32 bytes per lane, contiguous.
__device__ __forceinline__ v16bf load_b_frag(const bf16_t* fragbase, int tile, int lane) {
  const bf16_t* p = fragbase + ((size_t)(tile * 32 + lane)) * 16;
  return frag_from_chunks(p, p + 8);
}

__device__ __forceinline__ float gelu_exact(float x) {
  return 0.5f * x * (1.0f + erff(x * 0.70710678118654752f));
}

// nearest-neighbor gather: output row n in [0,19200) -> source row in [0,576)
__device__ __forceinline__ int raw_src_row(int n) {
  int h = n / W_;
  int w = n - h * W_;
  return (h / 5) * S_ + (w * 3) / 20;
}

// ---------------- prep kernels ----------------
// token_init [B,12,12,256] -> t_exp fp32+bf16 [B,576,256] (2x repeat_interleave)
__global__ void k_expand(const float* __restrict__ tok, float* __restrict__ texpf,
                         bf16_t* __restrict__ texpb) {
  int t = blockIdx.x * blockDim.x + threadIdx.x;
  if (t >= B_ * N576 * D_) return;
  int d = t & (D_ - 1);
  int r = (t >> 8) % N576;
  int b = (t >> 8) / N576;
  int y = r / S_, x = r - y * S_;
  float v = tok[((size_t)b * 144 + (y >> 1) * 12 + (x >> 1)) * D_ + d];
  texpf[t] = v;
  texpb[t] = (bf16_t)v;
}

__global__ void k_cvt_pt(const float* __restrict__ pt, bf16_t* __restrict__ ptb) {
  int t = blockIdx.x * blockDim.x + threadIdx.x;
  if (t < B_ * P_ * D_) ptb[t] = (bf16_t)pt[t];
}

// swizzle fp32 row-major KxN weight into bf16 WMMA B fragments
// frag layout: [kt][nt][lane][16], tile index = kt*(N/16)+nt
__global__ void k_swz_w(const float* __restrict__ w, bf16_t* __restrict__ frag,
                        int K, int N) {
  int t = blockIdx.x * blockDim.x + threadIdx.x;
  int ntn = N >> 4, ntk = K >> 5;
  int total = ntk * ntn * 32;
  if (t >= total) return;
  int lane = t & 31, tile = t >> 5;
  int nt = tile % ntn, kt = tile / ntn;
  int n  = nt * 16 + (lane & 15);
  int kb = kt * 32 + ((lane >> 4) << 4);
  bf16_t* o = frag + (size_t)t * 16;
#pragma unroll
  for (int j = 0; j < 16; ++j) o[j] = (bf16_t)w[(size_t)(kb + j) * N + n];
}

// ---------------- K1: adj0 = t_exp . pt^T / 16  (WMMA) ----------------
// grid (36, B), block 32 (one wave per 16-row tile)
__global__ void k_adj0(const bf16_t* __restrict__ texpb, const bf16_t* __restrict__ ptb,
                       bf16_t* __restrict__ adj0b) {
  int lane = threadIdx.x;
  int b = blockIdx.y, mt = blockIdx.x;
  int mm = (lane >> 4) * 8, nn = lane & 15;
  const bf16_t* Arow = texpb + ((size_t)b * N576 + mt * 16 + nn) * D_;
  v16bf a[8];
#pragma unroll
  for (int kt = 0; kt < 8; ++kt) a[kt] = load_a_frag(Arow, kt * 32, lane);
  // B[k][p] = pt[p][k]: lane's column row in pt is contiguous in k -> direct frag
  const bf16_t* ptbase = ptb + (size_t)b * P_ * D_ + (size_t)nn * D_ + ((lane >> 4) << 4);
  for (int nt = 0; nt < 8; ++nt) {
    v8f c;
#pragma unroll
    for (int r = 0; r < 8; ++r) c[r] = 0.0f;
#pragma unroll
    for (int kt = 0; kt < 8; ++kt) {
      const bf16_t* bp = ptbase + (size_t)nt * 16 * D_ + kt * 32;
      c = wmma_bf16(a[kt], frag_from_chunks(bp, bp + 8), c);
    }
#pragma unroll
    for (int r = 0; r < 8; ++r) {
      float v = c[r] * 0.0625f;  // dim^-0.5
      adj0b[((size_t)b * N576 + mt * 16 + mm + r) * P_ + nt * 16 + nn] = (bf16_t)v;
    }
  }
}

// ---------------- K2: nr MLP over 576 rows (WMMA, fused GELU) ----------------
// grid (36, B), block 32, 16KB LDS for hidden transpose
__global__ void k_mlp_nr(const bf16_t* __restrict__ adj0b,
                         const bf16_t* __restrict__ w1f, const float* __restrict__ b1,
                         const bf16_t* __restrict__ w2f, const float* __restrict__ b2,
                         float* __restrict__ postf) {
  __shared__ __align__(16) bf16_t hid[16 * PH];
  int lane = threadIdx.x;
  int b = blockIdx.y, mt = blockIdx.x;
  int mm = (lane >> 4) * 8, nn = lane & 15;
  const bf16_t* Arow = adj0b + ((size_t)b * N576 + mt * 16 + nn) * P_;
  v16bf a[4];
#pragma unroll
  for (int kt = 0; kt < 4; ++kt) a[kt] = load_a_frag(Arow, kt * 32, lane);
  for (int nt = 0; nt < 32; ++nt) {
    float bias = b1[nt * 16 + nn];
    v8f c;
#pragma unroll
    for (int r = 0; r < 8; ++r) c[r] = bias;
#pragma unroll
    for (int kt = 0; kt < 4; ++kt)
      c = wmma_bf16(a[kt], load_b_frag(w1f, kt * 32 + nt, lane), c);
#pragma unroll
    for (int r = 0; r < 8; ++r)
      hid[(mm + r) * PH + nt * 16 + nn] = (bf16_t)gelu_exact(c[r]);
  }
  __syncthreads();
  const bf16_t* Hrow = hid + nn * PH;
  for (int nt = 0; nt < 8; ++nt) {
    float bias = b2[nt * 16 + nn];
    v8f c;
#pragma unroll
    for (int r = 0; r < 8; ++r) c[r] = bias;
#pragma unroll
    for (int kt = 0; kt < 16; ++kt)
      c = wmma_bf16(load_a_frag(Hrow, kt * 32, lane),
                    load_b_frag(w2f, kt * 8 + nt, lane), c);
#pragma unroll
    for (int r = 0; r < 8; ++r)
      postf[((size_t)b * N576 + mt * 16 + mm + r) * P_ + nt * 16 + nn] = c[r];
  }
}

// ---------------- K3: node_w/node_h contraction (K=24, LDS-tiled VALU) --------
// token_n stored transposed: tn[b][q=p*256+d][s], s in [0,48)
// grid (48*4, B), block 256
__global__ void k_tn(const float* __restrict__ postf, const float* __restrict__ texpf,
                     float* __restrict__ tn) {
  __shared__ float adjs[24 * 128];
  __shared__ float txs[24 * 64];
  int b = blockIdx.y;
  int s  = blockIdx.x >> 2;
  int dc = blockIdx.x & 3;
  int tid = threadIdx.x;
  for (int i = tid; i < 24 * 128; i += 256) {
    int k = i >> 7, p = i & 127;
    int row = (s < S_) ? (s * S_ + k) : (k * S_ + (s - S_));
    adjs[i] = postf[((size_t)b * N576 + row) * P_ + p];
  }
  for (int i = tid; i < 24 * 64; i += 256) {
    int k = i >> 6, d = i & 63;
    int row = (s < S_) ? (s * S_ + k) : (k * S_ + (s - S_));
    txs[i] = texpf[((size_t)b * N576 + row) * D_ + dc * 64 + d];
  }
  __syncthreads();
  int p = tid & 127, ds0 = (tid >> 7) * 32;
  float av[24];
#pragma unroll
  for (int k = 0; k < 24; ++k) av[k] = adjs[k * 128 + p];
  const float isr = 0.2041241452319315f;  // 24^-0.5
  for (int dl = 0; dl < 32; ++dl) {
    int d = ds0 + dl;
    float acc = 0.f;
#pragma unroll
    for (int k = 0; k < 24; ++k) acc += av[k] * txs[k * 64 + d];
    tn[((size_t)b * QD + (size_t)p * D_ + dc * 64 + d) * NEW2 + s] = acc * isr;
  }
}

// ---------------- K4: tf MLP (48->48 GELU ->1), one thread per q ------------
// grid (128, B), block 256
__global__ void k_tf(const float* __restrict__ tn, const float* __restrict__ w1,
                     const float* __restrict__ b1, const float* __restrict__ w2,
                     const float* __restrict__ b2, float* __restrict__ tf) {
  __shared__ float w1s[48 * 48];
  __shared__ float b1s[48];
  __shared__ float w2s[48];
  int tid = threadIdx.x;
  for (int i = tid; i < 48 * 48; i += 256) w1s[i] = w1[i];
  if (tid < 48) { b1s[tid] = b1[tid]; w2s[tid] = w2[tid]; }
  __syncthreads();
  int b = blockIdx.y;
  int q = blockIdx.x * 256 + tid;
  const float* x = tn + ((size_t)b * QD + q) * NEW2;
  float xv[48];
#pragma unroll
  for (int c = 0; c < 48; ++c) xv[c] = x[c];
  float o = b2[0];
  for (int c2 = 0; c2 < 48; ++c2) {
    float s = b1s[c2];
#pragma unroll
    for (int c = 0; c < 48; ++c) s += xv[c] * w1s[c * 48 + c2];
    o += gelu_exact(s) * w2s[c2];
  }
  tf[(size_t)b * QD + q] = o;
}

// swizzle token_fused [B,128,256] into per-batch WMMA B fragments (K=128,N=256)
__global__ void k_swz_tf(const float* __restrict__ tf, bf16_t* __restrict__ frag) {
  int b = blockIdx.y;
  int t = blockIdx.x * blockDim.x + threadIdx.x;
  const int total = 4 * 16 * 32;  // ntk * ntn * lanes
  if (t >= total) return;
  int lane = t & 31, tile = t >> 5;
  int nt = tile & 15, kt = tile >> 4;
  int n  = nt * 16 + (lane & 15);
  int kb = kt * 32 + ((lane >> 4) << 4);
  bf16_t* o = frag + ((size_t)b * total + t) * 16;
#pragma unroll
  for (int j = 0; j < 16; ++j)
    o[j] = (bf16_t)tf[(size_t)b * QD + (size_t)(kb + j) * D_ + n];
}

// ---------------- K5: fused na MLP + softmax + attn@fused + residual -------
// One wave per 16 output rows: gather adj0 rows (token_raw scores reuse!),
// GEMM1(16x128x512)+GELU -> LDS -> GEMM2(16x512x128) -> softmax ->
// GEMM3(16x128x256) + residual -> d_out.  grid (1200, B), block 32.
__global__ void k_na_fused(const bf16_t* __restrict__ adj0b,
                           const bf16_t* __restrict__ w1f, const float* __restrict__ b1,
                           const bf16_t* __restrict__ w2f, const float* __restrict__ b2,
                           const bf16_t* __restrict__ tff,
                           const float* __restrict__ texpf, float* __restrict__ out) {
  __shared__ __align__(16) bf16_t hid[16 * PH];
  int lane = threadIdx.x;
  int b = blockIdx.y, t = blockIdx.x;
  int mm = (lane >> 4) * 8, nn = lane & 15;

  // gathered A rows (node_attn pre-MLP scores == gathered adj0 rows)
  const bf16_t* Arow = adj0b + ((size_t)b * N576 + raw_src_row(t * 16 + nn)) * P_;
  v16bf a[4];
#pragma unroll
  for (int kt = 0; kt < 4; ++kt) a[kt] = load_a_frag(Arow, kt * 32, lane);

  // GEMM1 + GELU -> LDS (hidden 16x512)
  for (int nt = 0; nt < 32; ++nt) {
    float bias = b1[nt * 16 + nn];
    v8f c;
#pragma unroll
    for (int r = 0; r < 8; ++r) c[r] = bias;
#pragma unroll
    for (int kt = 0; kt < 4; ++kt)
      c = wmma_bf16(a[kt], load_b_frag(w1f, kt * 32 + nt, lane), c);
#pragma unroll
    for (int r = 0; r < 8; ++r)
      hid[(mm + r) * PH + nt * 16 + nn] = (bf16_t)gelu_exact(c[r]);
  }
  __syncthreads();

  // GEMM2 -> scores [16x128] in D layout
  v8f sc[8];
  const bf16_t* Hrow = hid + nn * PH;
  for (int nt = 0; nt < 8; ++nt) {
    float bias = b2[nt * 16 + nn];
    v8f c;
#pragma unroll
    for (int r = 0; r < 8; ++r) c[r] = bias;
#pragma unroll
    for (int kt = 0; kt < 16; ++kt)
      c = wmma_bf16(load_a_frag(Hrow, kt * 32, lane),
                    load_b_frag(w2f, kt * 8 + nt, lane), c);
    sc[nt] = c;
  }

  // softmax over 128 per row: per-lane 8 partials + 16-lane xor-shuffle reduce
#pragma unroll
  for (int r = 0; r < 8; ++r) {
    float m1 = -3.4e38f;
#pragma unroll
    for (int nt = 0; nt < 8; ++nt) m1 = fmaxf(m1, sc[nt][r]);
    for (int d = 1; d < 16; d <<= 1) m1 = fmaxf(m1, __shfl_xor(m1, d, 32));
    float s = 0.f;
#pragma unroll
    for (int nt = 0; nt < 8; ++nt) { float e = __expf(sc[nt][r] - m1); sc[nt][r] = e; s += e; }
    for (int d = 1; d < 16; d <<= 1) s += __shfl_xor(s, d, 32);
    float inv = 1.0f / s;
#pragma unroll
    for (int nt = 0; nt < 8; ++nt) sc[nt][r] *= inv;
  }
  __syncthreads();

  // stage attn bf16 into LDS in row-major [16x128] for A-fragment reload
#pragma unroll
  for (int nt = 0; nt < 8; ++nt)
#pragma unroll
    for (int r = 0; r < 8; ++r)
      hid[(mm + r) * P_ + nt * 16 + nn] = (bf16_t)sc[nt][r];
  __syncthreads();

  // GEMM3: attn(16x128) @ token_fused(128x256) + token_raw residual
  const bf16_t* At = hid + nn * P_;
  v16bf a3[4];
#pragma unroll
  for (int kt = 0; kt < 4; ++kt) a3[kt] = load_a_frag(At, kt * 32, lane);
  const bf16_t* tfb = tff + (size_t)b * (4 * 16 * 32 * 16);
  int srow[8];
#pragma unroll
  for (int r = 0; r < 8; ++r) srow[r] = raw_src_row(t * 16 + mm + r);
  for (int nt = 0; nt < 16; ++nt) {
    v8f c;
#pragma unroll
    for (int r = 0; r < 8; ++r) c[r] = 0.0f;
#pragma unroll
    for (int kt = 0; kt < 4; ++kt)
      c = wmma_bf16(a3[kt], load_b_frag(tfb, kt * 16 + nt, lane), c);
    int d = nt * 16 + nn;
#pragma unroll
    for (int r = 0; r < 8; ++r) {
      int n = t * 16 + mm + r;
      float res = texpf[((size_t)b * N576 + srow[r]) * D_ + d];
      out[((size_t)b * HW_ + n) * D_ + d] = c[r] + res;
    }
  }
}

// ---------------- host launcher ----------------
extern "C" void kernel_launch(void* const* d_in, const int* in_sizes, int n_in,
                              void* d_out, int out_size, void* d_ws, size_t ws_size,
                              hipStream_t stream) {
  (void)in_sizes; (void)n_in; (void)out_size; (void)ws_size;
  const float* token_init  = (const float*)d_in[0];
  const float* point_token = (const float*)d_in[1];
  const float* nr_w1 = (const float*)d_in[2];
  const float* nr_b1 = (const float*)d_in[3];
  const float* nr_w2 = (const float*)d_in[4];
  const float* nr_b2 = (const float*)d_in[5];
  const float* na_w1 = (const float*)d_in[6];
  const float* na_b1 = (const float*)d_in[7];
  const float* na_w2 = (const float*)d_in[8];
  const float* na_b2 = (const float*)d_in[9];
  const float* tf_w1 = (const float*)d_in[10];
  const float* tf_b1 = (const float*)d_in[11];
  const float* tf_w2 = (const float*)d_in[12];
  const float* tf_b2 = (const float*)d_in[13];

  char* ws = (char*)d_ws;
  size_t off = 0;
  auto alloc = [&](size_t bytes) -> void* {
    void* p = ws + off;
    off += (bytes + 255) & ~(size_t)255;
    return p;
  };
  float*  texpf  = (float*) alloc((size_t)B_ * N576 * D_ * 4);   // 2.36 MB
  bf16_t* texpb  = (bf16_t*)alloc((size_t)B_ * N576 * D_ * 2);   // 1.18 MB
  bf16_t* ptb    = (bf16_t*)alloc((size_t)B_ * P_ * D_ * 2);     // 0.26 MB
  bf16_t* adj0b  = (bf16_t*)alloc((size_t)B_ * N576 * P_ * 2);   // 0.59 MB
  float*  postf  = (float*) alloc((size_t)B_ * N576 * P_ * 4);   // 1.18 MB
  float*  tn     = (float*) alloc((size_t)B_ * QD * NEW2 * 4);   // 25.2 MB
  float*  tff32  = (float*) alloc((size_t)B_ * QD * 4);          // 0.52 MB
  bf16_t* tffrag = (bf16_t*)alloc((size_t)B_ * 4 * 16 * 32 * 16 * 2);
  bf16_t* w1f_nr = (bf16_t*)alloc((size_t)4  * 32 * 32 * 16 * 2);  // 128KB each
  bf16_t* w2f_nr = (bf16_t*)alloc((size_t)16 * 8  * 32 * 16 * 2);
  bf16_t* w1f_na = (bf16_t*)alloc((size_t)4  * 32 * 32 * 16 * 2);
  bf16_t* w2f_na = (bf16_t*)alloc((size_t)16 * 8  * 32 * 16 * 2);

  // prep
  k_expand<<<(B_ * N576 * D_ + 255) / 256, 256, 0, stream>>>(token_init, texpf, texpb);
  k_cvt_pt<<<(B_ * P_ * D_ + 255) / 256, 256, 0, stream>>>(point_token, ptb);
  k_swz_w<<<16, 256, 0, stream>>>(nr_w1, w1f_nr, P_, PH);   // 4096 threads
  k_swz_w<<<16, 256, 0, stream>>>(nr_w2, w2f_nr, PH, P_);
  k_swz_w<<<16, 256, 0, stream>>>(na_w1, w1f_na, P_, PH);
  k_swz_w<<<16, 256, 0, stream>>>(na_w2, w2f_na, PH, P_);

  // adj0 = t_exp . pt^T * dim^-0.5   (also the gathered token_raw scores)
  k_adj0<<<dim3(N576 / 16, B_), 32, 0, stream>>>(texpb, ptb, adj0b);

  // node_adj MLP (nr)
  k_mlp_nr<<<dim3(N576 / 16, B_), 32, 0, stream>>>(adj0b, w1f_nr, nr_b1,
                                                   w2f_nr, nr_b2, postf);
  // token_n (node_w ++ node_h), transposed to [B, 32768, 48]
  k_tn<<<dim3(NEW2 * 4, B_), 256, 0, stream>>>(postf, texpf, tn);

  // token_fused MLP + swizzle to WMMA fragments
  k_tf<<<dim3(QD / 256, B_), 256, 0, stream>>>(tn, tf_w1, tf_b1, tf_w2, tf_b2, tff32);
  k_swz_tf<<<dim3((4 * 16 * 32 + 63) / 64, B_), 64, 0, stream>>>(tff32, tffrag);

  // fused: gather scores -> MLP -> softmax -> @token_fused -> +token_raw
  k_na_fused<<<dim3(HW_ / 16, B_), 32, 0, stream>>>(adj0b, w1f_na, na_b1,
                                                    w2f_na, na_b2, tffrag,
                                                    texpf, (float*)d_out);
}
